// FeatureAdaption_26259430048581
// MI455X (gfx1250) — compile-verified, matrix operand order checked by reading, not run on hardware
//
#include <hip/hip_runtime.h>

typedef __attribute__((ext_vector_type(16))) _Float16 v16h;
typedef __attribute__((ext_vector_type(8)))  _Float16 v8h;
typedef __attribute__((ext_vector_type(8)))  float    v8f;

// Problem constants (B=2, C=256, H=W=128, O=256, K=3, DG=4)
#define HWv   16384
#define Hdim  128
#define Wdim  128
#define Cdim  256
#define Odim  256
#define OCh   72          // DG*K*K*2 offset channels
#define KTOT  2304        // C*9
#define KBLKS 72          // KTOT/32
#define NBLKS 16          // Odim/16

// Workspace layout (bytes)
static constexpr size_t OFF_BYTES = (size_t)2 * OCh * HWv * 4;          // 9,437,184
static constexpr size_t WO_ELEMS  = (size_t)8 * 5 * 32 * 16;            // 20,480 halves
static constexpr size_t WO_BYTES  = WO_ELEMS * 2;

__device__ __forceinline__ float relu(float v) { return v > 0.f ? v : 0.f; }

// ---------------------------------------------------------------------------
// Pack w_deform [O=256][C=256][3][3] fp32 -> f16 fragment-ready B layout:
//   out[((kblk*16 + nblk)*32 + lane)*16 + e], k = c*9 + tap
//   lane&15 -> n within block; kb=(lane>>4)*8; e<8 -> k=kb+e, e>=8 -> k=kb+16+(e-8)
// ---------------------------------------------------------------------------
__global__ __launch_bounds__(256) void pack_wdef(const float* __restrict__ wd,
                                                 _Float16* __restrict__ out) {
  int tid = blockIdx.x * 256 + threadIdx.x;            // 589,824 total
  int e    = tid & 15;
  int lane = (tid >> 4) & 31;
  int nblk = (tid >> 9) & 15;
  int kblk = tid >> 13;
  int o  = nblk * 16 + (lane & 15);
  int kb = (lane >> 4) * 8;
  int kl = (e < 8) ? (kb + e) : (kb + 16 + (e - 8));
  int k  = kblk * 32 + kl;
  int c  = k / 9;
  int tap = k - c * 9;
  out[tid] = (_Float16)wd[(o * Cdim + c) * 9 + tap];
}

// Pack w_offset [72][256] fp32 -> f16 fragment-ready layout, N padded 72->80
__global__ __launch_bounds__(256) void pack_woff(const float* __restrict__ wo,
                                                 _Float16* __restrict__ out) {
  int tid = blockIdx.x * 256 + threadIdx.x;            // 20,480 total
  int e    = tid & 15;
  int lane = (tid >> 4) & 31;
  int rest = tid >> 9;
  int nblk = rest % 5;
  int kblk = rest / 5;
  int oc = nblk * 16 + (lane & 15);
  int kb = (lane >> 4) * 8;
  int kl = (e < 8) ? (kb + e) : (kb + 16 + (e - 8));
  int k  = kblk * 32 + kl;
  out[tid] = (oc < OCh) ? (_Float16)wo[oc * Cdim + k] : (_Float16)0.0f;
}

// ---------------------------------------------------------------------------
// Kernel 1: offset = x (as [M=BHW, K=256]) @ w_offset^T + bias  via f16 WMMA
// Block: 256 thr, 128 pixels, 8 waves each own one 16-row M-block x 80 cols.
// Double-buffered A tile in LDS, one barrier per K-step.
// ---------------------------------------------------------------------------
__global__ __launch_bounds__(256) void offset_gemm(const float* __restrict__ x,
                                                   const _Float16* __restrict__ wopk,
                                                   const float* __restrict__ bias,
                                                   float* __restrict__ off_buf) {
  __shared__ __align__(16) _Float16 a_lds[2][128][40];  // 20.4 KB
  const int t    = threadIdx.x;
  const int lane = t & 31;
  const int wave = t >> 5;
  const int p0   = blockIdx.x * 128;
  const int b    = p0 >> 14;            // HW = 16384
  const int hw0  = p0 & (HWv - 1);

  const int row  = t & 127;             // staging: pixel row
  const int half = t >> 7;              // staging: channel half (0/1)
  const float* xcol = x + (size_t)b * Cdim * HWv + hw0 + row;

  // stage first 128x32 A tile into buffer 0
  {
    v8h va, vb;
#pragma unroll
    for (int j = 0; j < 8; ++j)  va[j] = (_Float16)xcol[(size_t)(half * 16 + j) * HWv];
#pragma unroll
    for (int j = 0; j < 8; ++j)  vb[j] = (_Float16)xcol[(size_t)(half * 16 + 8 + j) * HWv];
    *(v8h*)&a_lds[0][row][half * 16]     = va;
    *(v8h*)&a_lds[0][row][half * 16 + 8] = vb;
  }
  __syncthreads();

  v8f acc[5] = {};
  const int kb = (lane >> 4) * 8;
  const int aRowIdx = wave * 16 + (lane & 15);

  for (int kblk = 0; kblk < 8; ++kblk) {
    const _Float16* aRow = &a_lds[kblk & 1][aRowIdx][0];
    v8h lo = *(const v8h*)(aRow + kb);
    v8h hi = *(const v8h*)(aRow + kb + 16);
    v16h a = __builtin_shufflevector(lo, hi, 0,1,2,3,4,5,6,7,8,9,10,11,12,13,14,15);

    // preload all 5 B fragments, then run WMMAs back-to-back
    v16h bfr[5];
#pragma unroll
    for (int nb = 0; nb < 5; ++nb)
      bfr[nb] = *(const v16h*)(wopk + ((size_t)(kblk * 5 + nb) * 32 + lane) * 16);
#pragma unroll
    for (int nb = 0; nb < 5; ++nb)
      acc[nb] = __builtin_amdgcn_wmma_f32_16x16x32_f16(false, a, false, bfr[nb],
                                                       (short)0, acc[nb], false, false);

    // stage next K-step into other buffer
    if (kblk + 1 < 8) {
      const float* xn = xcol + (size_t)(kblk + 1) * 32 * HWv;
      v8h va, vb;
#pragma unroll
      for (int j = 0; j < 8; ++j)  va[j] = (_Float16)xn[(size_t)(half * 16 + j) * HWv];
#pragma unroll
      for (int j = 0; j < 8; ++j)  vb[j] = (_Float16)xn[(size_t)(half * 16 + 8 + j) * HWv];
      *(v8h*)&a_lds[(kblk + 1) & 1][row][half * 16]     = va;
      *(v8h*)&a_lds[(kblk + 1) & 1][row][half * 16 + 8] = vb;
    }
    __syncthreads();
  }

  // store with fused bias; C/D layout: VGPR r -> M=r (lanes 0-15), r+8 (16-31)
  const int rowBase = wave * 16 + ((lane >> 4) << 3);
#pragma unroll
  for (int nb = 0; nb < 5; ++nb) {
    int n = nb * 16 + (lane & 15);
    if (n < OCh) {
      float bv = bias[n];
      float* col = off_buf + (size_t)(b * OCh + n) * HWv + hw0 + rowBase;
      float4 s0 = make_float4(acc[nb][0] + bv, acc[nb][1] + bv,
                              acc[nb][2] + bv, acc[nb][3] + bv);
      float4 s1 = make_float4(acc[nb][4] + bv, acc[nb][5] + bv,
                              acc[nb][6] + bv, acc[nb][7] + bv);
      *(float4*)(col)     = s0;
      *(float4*)(col + 4) = s1;
    }
  }
}

// ---------------------------------------------------------------------------
// Kernel 2: deformable conv main GEMM with fused bilinear im2col.
// Block: 256 thr, 64 pixels, full N=256. Waves: mblk = wave>>1, nhalf = wave&1.
// A tile double-buffered in LDS; thread owns 8 contiguous k of one row
// => one ds_store_b128 per K-step, one divide-by-9 per K-step.
// ---------------------------------------------------------------------------
__global__ __launch_bounds__(256) void deform_main(const float* __restrict__ x,
                                                   const float* __restrict__ off_buf,
                                                   const _Float16* __restrict__ wdpk,
                                                   float* __restrict__ out) {
  __shared__ int    coord_i[2304];                    // packed (y0,x0) int16 pair
  __shared__ float2 coord_l[2304];                    // (ly, lx)
  __shared__ __align__(16) _Float16 a_lds[2][64][40]; // 10.2 KB double buffer

  const int t    = threadIdx.x;
  const int lane = t & 31;
  const int wave = t >> 5;
  const int p0   = blockIdx.x * 64;
  const int b    = p0 >> 14;
  const int hw0  = p0 & (HWv - 1);
  const float* xb = x + (size_t)b * Cdim * HWv;

  // ---- stage sampling coords: (pixel m, group g, tap) -> y0,x0,ly,lx -------
#pragma unroll
  for (int i = 0; i < 9; ++i) {
    int idx = i * 256 + t;             // = m*36 + g*9 + tap, 2304 total
    int m   = idx / 36;
    int rem = idx - m * 36;
    int g   = rem / 9;
    int tap = rem - g * 9;
    int hw  = hw0 + m;
    int h = hw >> 7, w = hw & 127;
    const float* ob = off_buf + (size_t)b * OCh * HWv + (size_t)(g * 18 + tap * 2) * HWv + hw;
    float dy = ob[0];
    float dx = ob[HWv];
    float ys = (float)(h - 1 + tap / 3) + dy;
    float xs = (float)(w - 1 + tap % 3) + dx;
    float y0f = floorf(ys), x0f = floorf(xs);
    int y0 = (int)y0f, x0 = (int)x0f;
    coord_i[idx] = (y0 & 0xffff) | (x0 << 16);
    coord_l[idx] = make_float2(ys - y0f, xs - x0f);
  }

  // staging identity: thread owns row m = t>>2, k-span q*8..q*8+7
  const int sm = t >> 2;
  const int sq = (t & 3) * 8;

  // stage A tile for kblk=0 into buffer 0 (coords just written by this block;
  // barrier below covers both coords and first tile)
  auto stage = [&](int kblk, int buf) {
    const int k0 = kblk * 32 + sq;
    const int c0 = k0 / 9;
    const int tap0 = k0 - c0 * 9;
    v8h vals;
#pragma unroll
    for (int j = 0; j < 8; ++j) {
      int tap = tap0 + j;
      int wrap = tap >= 9;
      int c = c0 + wrap;
      tap -= wrap * 9;
      int g = c >> 6;                  // 64 channels per deform group
      int idx = sm * 36 + g * 9 + tap;
      int ci = coord_i[idx];
      float2 l = coord_l[idx];
      int y0 = (short)(ci & 0xffff);
      int x0 = (short)(ci >> 16);
      float ly = l.x, lx = l.y;
      const float* plane = xb + (size_t)c * HWv;
      int y1 = y0 + 1, x1 = x0 + 1;
      float fy0 = (float)((y0 >= 0) & (y0 < Hdim));
      float fy1 = (float)((y1 >= 0) & (y1 < Hdim));
      float fx0 = (float)((x0 >= 0) & (x0 < Wdim));
      float fx1 = (float)((x1 >= 0) & (x1 < Wdim));
      int yc0 = min(max(y0, 0), Hdim - 1), yc1 = min(max(y1, 0), Hdim - 1);
      int xc0 = min(max(x0, 0), Wdim - 1), xc1 = min(max(x1, 0), Wdim - 1);
      float v00 = plane[yc0 * Wdim + xc0] * (fy0 * fx0);
      float v01 = plane[yc0 * Wdim + xc1] * (fy0 * fx1);
      float v10 = plane[yc1 * Wdim + xc0] * (fy1 * fx0);
      float v11 = plane[yc1 * Wdim + xc1] * (fy1 * fx1);
      float val = (1.f - ly) * ((1.f - lx) * v00 + lx * v01)
                +        ly  * ((1.f - lx) * v10 + lx * v11);
      vals[j] = (_Float16)val;
    }
    *(v8h*)&a_lds[buf][sm][sq] = vals;  // single ds_store_b128
  };

  __syncthreads();          // coords visible
  stage(0, 0);
  __syncthreads();          // first tile visible

  v8f acc[8] = {};
  const int mblk  = wave >> 1;
  const int nhalf = wave & 1;
  const int kb    = (lane >> 4) * 8;
  const int aRowIdx = mblk * 16 + (lane & 15);

  for (int kblk = 0; kblk < KBLKS; ++kblk) {
    // A fragment: two ds_load_b128 per documented 16-bit A striping
    const _Float16* aRow = &a_lds[kblk & 1][aRowIdx][0];
    v8h lo = *(const v8h*)(aRow + kb);
    v8h hi = *(const v8h*)(aRow + kb + 16);
    v16h a = __builtin_shufflevector(lo, hi, 0,1,2,3,4,5,6,7,8,9,10,11,12,13,14,15);

    const _Float16* bbase = wdpk + ((size_t)(kblk * NBLKS + nhalf * 8) * 32 + lane) * 16;
    if (kblk + 1 < KBLKS)  // global_prefetch_b8 for next K-step's B fragments
      __builtin_prefetch(wdpk + ((size_t)((kblk + 1) * NBLKS + nhalf * 8) * 32 + lane) * 16, 0, 1);

    // B fragments preloaded in groups of 4: clause the loads, batch the WMMAs
    v16h bfr[4];
#pragma unroll
    for (int f = 0; f < 4; ++f) bfr[f] = *(const v16h*)(bbase + (size_t)f * 512);
#pragma unroll
    for (int f = 0; f < 4; ++f)
      acc[f] = __builtin_amdgcn_wmma_f32_16x16x32_f16(false, a, false, bfr[f],
                                                      (short)0, acc[f], false, false);
#pragma unroll
    for (int f = 0; f < 4; ++f) bfr[f] = *(const v16h*)(bbase + (size_t)(f + 4) * 512);
#pragma unroll
    for (int f = 0; f < 4; ++f)
      acc[f + 4] = __builtin_amdgcn_wmma_f32_16x16x32_f16(false, a, false, bfr[f],
                                                          (short)0, acc[f + 4], false, false);

    // stage next K-step into the other buffer (overlaps with WMMA execution)
    if (kblk + 1 < KBLKS) stage(kblk + 1, (kblk + 1) & 1);
    __syncthreads();
  }

  // ---- fused ReLU store (two float4 per accumulator, 32B aligned) ---------
  float* ob = out + (size_t)b * Odim * HWv;
  const int rowBase = mblk * 16 + ((lane >> 4) << 3);
#pragma unroll
  for (int f = 0; f < 8; ++f) {
    int n = (nhalf * 8 + f) * 16 + (lane & 15);
    float* col = ob + (size_t)n * HWv + hw0 + rowBase;
    float4 s0 = make_float4(relu(acc[f][0]), relu(acc[f][1]),
                            relu(acc[f][2]), relu(acc[f][3]));
    float4 s1 = make_float4(relu(acc[f][4]), relu(acc[f][5]),
                            relu(acc[f][6]), relu(acc[f][7]));
    *(float4*)(col)     = s0;
    *(float4*)(col + 4) = s1;
  }
}

// ---------------------------------------------------------------------------
extern "C" void kernel_launch(void* const* d_in, const int* in_sizes, int n_in,
                              void* d_out, int out_size, void* d_ws, size_t ws_size,
                              hipStream_t stream) {
  const float* x        = (const float*)d_in[0];   // [2,256,128,128]
  const float* w_offset = (const float*)d_in[1];   // [72,256,1,1]
  const float* b_offset = (const float*)d_in[2];   // [72]
  const float* w_deform = (const float*)d_in[3];   // [256,256,3,3]
  float* out = (float*)d_out;

  char* ws = (char*)d_ws;
  float*    off_buf = (float*)ws;
  _Float16* wo_pack = (_Float16*)(ws + OFF_BYTES);
  _Float16* wd_pack = (_Float16*)(ws + OFF_BYTES + WO_BYTES);

  hipLaunchKernelGGL(pack_woff, dim3(80),   dim3(256), 0, stream, w_offset, wo_pack);
  hipLaunchKernelGGL(pack_wdef, dim3(2304), dim3(256), 0, stream, w_deform, wd_pack);
  hipLaunchKernelGGL(offset_gemm, dim3(256), dim3(256), 0, stream, x, wo_pack, b_offset, off_buf);
  hipLaunchKernelGGL(deform_main, dim3(512), dim3(256), 0, stream, x, off_buf, wd_pack, out);
}
